// GATNet_7670811591305
// MI455X (gfx1250) — matrix operand order repaired and checked by reading, not compiled, not run on hardware
//
#include <hip/hip_runtime.h>
#include <hip/hip_bf16.h>
#include <stdint.h>
#include <stddef.h>

#define NN      50000
#define GG      256
#define HC      200
#define CPH     100
#define NTILES  13      // ceil(200/16)
#define MTILES  3125    // 50000/16 exactly
#define SLOPE   0.2f

typedef __attribute__((ext_vector_type(16))) __bf16 v16bf;
typedef __attribute__((ext_vector_type(8)))  __bf16 v8bf;
typedef __attribute__((ext_vector_type(8)))  float  v8f;

// ---------------------------------------------------------------- helpers
static __device__ __forceinline__ __bf16 f2bf(float f) {
  uint32_t u = __float_as_uint(f);
  u += 0x7FFFu + ((u >> 16) & 1u);          // round-to-nearest-even
  uint16_t h = (uint16_t)(u >> 16);
  return __builtin_bit_cast(__bf16, h);
}

static __device__ __forceinline__ void atomicMaxF(float* a, float v) {
  if (v >= 0.0f) atomicMax((int*)a, __float_as_int(v));
  else           atomicMin((unsigned int*)a, (unsigned int)__float_as_int(v));
}

// ---------------------------------------------------------------- converts
__global__ void cvt_x_bf16(const float* __restrict__ X, __bf16* __restrict__ XB,
                           int F, int Kpad, int total) {
  int gid = blockIdx.x * blockDim.x + threadIdx.x;
  if (gid >= total) return;
  int n = gid / Kpad, k = gid - n * Kpad;
  XB[gid] = (k < F) ? f2bf(X[(size_t)n * F + k]) : f2bf(0.0f);
}

// W is [K,200] row-major; produce WT [208][Kpad] bf16 (column-major W, zero-padded)
__global__ void cvt_w_bf16(const float* __restrict__ W, __bf16* __restrict__ WT,
                           int K, int Kpad) {
  int gid = blockIdx.x * blockDim.x + threadIdx.x;
  int total = NTILES * 16 * Kpad;
  if (gid >= total) return;
  int n = gid / Kpad, k = gid - n * Kpad;
  WT[gid] = (n < HC && k < K) ? f2bf(W[(size_t)k * HC + n]) : f2bf(0.0f);
}

// ---------------------------------------------------------------- WMMA GEMM
// H[m, n] = sum_k A_bf16[m, k] * W[k, n]   (A: [50000, Kpad] bf16, WT: [208, Kpad] bf16)
// One wave computes a 16-row stripe across all 13 N-tiles.
// Software-pipelined: A fragment for k-step ks+1 and B fragment for tile t+1 are
// issued before the WMMA that consumes the current fragments, so the compiler
// emits partial s_wait_loadcnt instead of full drains.
__global__ void __launch_bounds__(256)
gemm_bf16_wmma(const __bf16* __restrict__ A, const __bf16* __restrict__ Bt,
               float* __restrict__ Hout, int Kpad) {
  const int wave  = threadIdx.x >> 5;
  const int lane  = threadIdx.x & 31;
  const int mtile = blockIdx.x * (blockDim.x >> 5) + wave;
  if (mtile >= MTILES) return;                 // wave-uniform: EXEC stays all-ones
  const int half = lane >> 4;                  // 0: lanes 0-15, 1: lanes 16-31
  const int col  = lane & 15;

  v8f acc[NTILES];
#pragma unroll
  for (int t = 0; t < NTILES; ++t) acc[t] = v8f{0.f,0.f,0.f,0.f,0.f,0.f,0.f,0.f};

  // A-fragment: lane holds row (lane&15); elems 0-7 = K kb+8*half.., elems 8-15 = K kb+16+8*half..
  const __bf16* arow  = A + (size_t)(mtile * 16 + col) * Kpad;
  // B-fragment: lane holds column (t*16 + col); elems 0-7 = K kb+8*half.., elems 8-15 = K kb+16+8*half..
  const __bf16* bbase = Bt + (size_t)col * Kpad + 8 * half;
  const size_t  btile = (size_t)16 * Kpad;     // stride between N-tiles
  const int ksteps = Kpad >> 5;

  // ---- prologue: A fragment for ks=0
  v8bf alo = *(const v8bf*)(arow + 8 * half);
  v8bf ahi = *(const v8bf*)(arow + 16 + 8 * half);

  for (int ks = 0; ks < ksteps; ++ks) {
    const int kb = ks << 5;
    v16bf amat = __builtin_shufflevector(alo, ahi,
        0,1,2,3,4,5,6,7,8,9,10,11,12,13,14,15);
    // issue next k-step's A loads early (hidden under 13 WMMAs)
    if (ks + 1 < ksteps) {
      alo = *(const v8bf*)(arow + kb + 32 + 8 * half);
      ahi = *(const v8bf*)(arow + kb + 48 + 8 * half);
      __builtin_prefetch(arow + kb + 64, 0, 3);
    }
    // ---- B double buffer: preload tile 0
    const __bf16* bp = bbase + kb;
    v8bf blo = *(const v8bf*)(bp);
    v8bf bhi = *(const v8bf*)(bp + 16);
#pragma unroll
    for (int t = 0; t < NTILES; ++t) {
      v8bf nlo, nhi;
      if (t + 1 < NTILES) {                    // issue t+1 loads before wmma(t)
        const __bf16* np = bp + (t + 1) * btile;
        nlo = *(const v8bf*)(np);
        nhi = *(const v8bf*)(np + 16);
      }
      v16bf bmat = __builtin_shufflevector(blo, bhi,
          0,1,2,3,4,5,6,7,8,9,10,11,12,13,14,15);
      acc[t] = __builtin_amdgcn_wmma_f32_16x16x32_bf16(
          false, amat, false, bmat, (short)0, acc[t], false, false);
      blo = nlo; bhi = nhi;
    }
  }
  // C/D layout: VGPR r -> M = r + 8*half ; N = lane&15
#pragma unroll
  for (int t = 0; t < NTILES; ++t) {
    int n = t * 16 + col;
    if (n < HC) {
#pragma unroll
      for (int r = 0; r < 8; ++r) {
        int m = mtile * 16 + r + 8 * half;
        Hout[(size_t)m * HC + n] = acc[t][r];
      }
    }
  }
}

// ---------------------------------------------------------------- attention scores
// es[n,h] = dot(h[n,h,:], asrc[h,:])  (wave per node, shuffle reduce)
__global__ void scores_kernel(const float* __restrict__ Hh,
                              const float* __restrict__ asrc,
                              const float* __restrict__ adst,
                              float* __restrict__ ES, float* __restrict__ ED) {
  int wave = threadIdx.x >> 5, lane = threadIdx.x & 31;
  int n = blockIdx.x * (blockDim.x >> 5) + wave;
  if (n >= NN) return;
  float es0 = 0.f, es1 = 0.f, ed0 = 0.f, ed1 = 0.f;
  for (int j = lane; j < HC; j += 32) {
    float v = Hh[(size_t)n * HC + j];
    float s = v * asrc[j], d = v * adst[j];
    if (j < CPH) { es0 += s; ed0 += d; } else { es1 += s; ed1 += d; }
  }
#pragma unroll
  for (int m = 16; m >= 1; m >>= 1) {
    es0 += __shfl_xor(es0, m, 32);
    es1 += __shfl_xor(es1, m, 32);
    ed0 += __shfl_xor(ed0, m, 32);
    ed1 += __shfl_xor(ed1, m, 32);
  }
  if (lane == 0) {
    ES[n * 2 + 0] = es0; ES[n * 2 + 1] = es1;
    ED[n * 2 + 0] = ed0; ED[n * 2 + 1] = ed1;
  }
}

// ---------------------------------------------------------------- init kernels
__global__ void fill_f32(float* p, float v, int n) {
  int g = blockIdx.x * blockDim.x + threadIdx.x;
  if (g < n) p[g] = v;
}
__global__ void fill_neginf(float* p, int n) {
  int g = blockIdx.x * blockDim.x + threadIdx.x;
  if (g < n) p[g] = __uint_as_float(0xFF800000u);
}

// ---------------------------------------------------------------- edge passes
__global__ void edge_max_kernel(const int* __restrict__ ei, int E, int Etot,
                                const float* __restrict__ ES,
                                const float* __restrict__ ED,
                                float* __restrict__ MMAX) {
  int gid = blockIdx.x * blockDim.x + threadIdx.x;
  if (gid >= Etot * 2) return;
  int e = gid >> 1, h = gid & 1;
  int s, d;
  if (e < E) { s = ei[e]; d = ei[E + e]; } else { s = d = e - E; }
  float v = ES[s * 2 + h] + ED[d * 2 + h];
  v = v > 0.f ? v : SLOPE * v;                  // leaky relu
  atomicMaxF(&MMAX[d * 2 + h], v);
}

__global__ void edge_exp_kernel(const int* __restrict__ ei, int E, int Etot,
                                const float* __restrict__ ES,
                                const float* __restrict__ ED,
                                const float* __restrict__ MMAX,
                                float* __restrict__ SSUM,
                                float* __restrict__ P) {
  int gid = blockIdx.x * blockDim.x + threadIdx.x;
  if (gid >= Etot * 2) return;
  int e = gid >> 1, h = gid & 1;
  int s, d;
  if (e < E) { s = ei[e]; d = ei[E + e]; } else { s = d = e - E; }
  float v = ES[s * 2 + h] + ED[d * 2 + h];
  v = v > 0.f ? v : SLOPE * v;
  float p = __expf(v - MMAX[d * 2 + h]);
  P[gid] = p;                                   // gid == e*2+h
  atomicAdd(&SSUM[d * 2 + h], p);
}

__global__ void edge_norm_kernel(const int* __restrict__ ei, int E, int Etot,
                                 const float* __restrict__ SSUM,
                                 float* __restrict__ P) {
  int gid = blockIdx.x * blockDim.x + threadIdx.x;
  if (gid >= Etot * 2) return;
  int e = gid >> 1, h = gid & 1;
  int d = (e < E) ? ei[E + e] : (e - E);
  P[gid] = P[gid] / (SSUM[d * 2 + h] + 1e-16f);
}

// one block per edge, thread = channel; out[dst] += alpha * h[src]
__global__ void edge_aggr_kernel(const int* __restrict__ ei, int E,
                                 const float* __restrict__ P,
                                 const float* __restrict__ Hh,
                                 float* __restrict__ Out) {
  int e = blockIdx.x;
  int c = threadIdx.x;
  if (c >= HC) return;
  int s, d;
  if (e < E) { s = ei[e]; d = ei[E + e]; } else { s = d = e - E; }
  int h = c / CPH;
  float a = P[e * 2 + h];
  atomicAdd(&Out[(size_t)d * HC + c], a * Hh[(size_t)s * HC + c]);
}

__global__ void bias_relu_kernel(float* __restrict__ X, const float* __restrict__ b,
                                 int total) {
  int gid = blockIdx.x * blockDim.x + threadIdx.x;
  if (gid >= total) return;
  int c = gid % HC;
  float v = X[gid] + b[c];
  X[gid] = v > 0.f ? v : 0.f;
}

// ---------------------------------------------------------------- pool + MLP
__global__ void pool_kernel(const float* __restrict__ X, const int* __restrict__ batch,
                            float* __restrict__ POOL, float* __restrict__ CNT) {
  int gid = blockIdx.x * blockDim.x + threadIdx.x;
  if (gid >= NN * HC) return;
  int n = gid / HC, c = gid - n * HC;
  int g = batch[n];
  atomicAdd(&POOL[g * HC + c], X[gid]);
  if (c == 0) atomicAdd(&CNT[g], 1.0f);
}

__global__ void mlp_kernel(const float* __restrict__ In, const float* __restrict__ Wm,
                           const float* __restrict__ b, float* __restrict__ Outp,
                           int din, int dout, int do_relu,
                           const float* __restrict__ CNT) {
  int g = blockIdx.x;
  int j = threadIdx.x;
  if (j >= dout) return;
  float inv = 1.0f;
  if (CNT) { float c = CNT[g]; inv = 1.0f / (c > 1.0f ? c : 1.0f); }
  float acc = b[j];
  for (int i = 0; i < din; ++i)
    acc += In[(size_t)g * din + i] * inv * Wm[(size_t)i * dout + j];
  if (do_relu && acc < 0.f) acc = 0.f;
  Outp[(size_t)g * dout + j] = acc;
}

// ---------------------------------------------------------------- launcher
extern "C" void kernel_launch(void* const* d_in, const int* in_sizes, int n_in,
                              void* d_out, int out_size, void* d_ws, size_t ws_size,
                              hipStream_t stream) {
  const float* x     = (const float*)d_in[0];
  const int*   ei    = (const int*)d_in[1];
  const int*   batch = (const int*)d_in[2];
  const int E    = in_sizes[1] / 2;
  const int Etot = E + NN;
  const int KP1  = 352;                 // 336 padded to x32
  const int KP2  = 224;                 // 200 padded to x32

  char* base = (char*)d_ws;
  size_t off = 0;
  auto take = [&](size_t bytes) -> void* {
    void* p = base + off;
    off += (bytes + 255) & ~(size_t)255;
    return p;
  };
  __bf16* XBF  = (__bf16*)take((size_t)NN * KP1 * sizeof(__bf16));       // 35.2 MB
  __bf16* WT   = (__bf16*)take((size_t)NTILES * 16 * KP1 * sizeof(__bf16));
  float*  HB   = (float*)take((size_t)NN * HC * sizeof(float));          // 40 MB
  float*  XB   = (float*)take((size_t)NN * HC * sizeof(float));          // 40 MB
  float*  ES   = (float*)take((size_t)NN * 2 * sizeof(float));
  float*  ED   = (float*)take((size_t)NN * 2 * sizeof(float));
  float*  MM   = (float*)take((size_t)NN * 2 * sizeof(float));
  float*  SS   = (float*)take((size_t)NN * 2 * sizeof(float));
  float*  P    = (float*)take((size_t)Etot * 2 * sizeof(float));
  float*  POOL = (float*)take((size_t)GG * HC * sizeof(float));
  float*  CNT  = (float*)take((size_t)GG * sizeof(float));
  float*  M1   = (float*)take((size_t)GG * 100 * sizeof(float));
  float*  M2   = (float*)take((size_t)GG * 100 * sizeof(float));
  (void)ws_size; (void)n_in; (void)out_size;

  const int BT = 256;
  auto blocks = [](long long n, int b) { return (unsigned)((n + b - 1) / b); };

  const float* xin = x;
  int F = 336, Kpad = KP1;
  for (int L = 0; L < 5; ++L) {
    const float* W  = (const float*)d_in[3 + 4 * L];
    const float* as = (const float*)d_in[4 + 4 * L];
    const float* ad = (const float*)d_in[5 + 4 * L];
    const float* bc = (const float*)d_in[6 + 4 * L];

    cvt_x_bf16<<<blocks((long long)NN * Kpad, BT), BT, 0, stream>>>(
        xin, XBF, F, Kpad, NN * Kpad);
    cvt_w_bf16<<<blocks((long long)NTILES * 16 * Kpad, BT), BT, 0, stream>>>(
        W, WT, F, Kpad);
    gemm_bf16_wmma<<<blocks(MTILES, 8), 256, 0, stream>>>(XBF, WT, HB, Kpad);

    scores_kernel<<<blocks(NN, 8), 256, 0, stream>>>(HB, as, ad, ES, ED);
    fill_neginf<<<blocks(NN * 2, BT), BT, 0, stream>>>(MM, NN * 2);
    fill_f32<<<blocks(NN * 2, BT), BT, 0, stream>>>(SS, 0.f, NN * 2);
    fill_f32<<<blocks((long long)NN * HC, BT), BT, 0, stream>>>(XB, 0.f, NN * HC);

    edge_max_kernel<<<blocks((long long)Etot * 2, BT), BT, 0, stream>>>(
        ei, E, Etot, ES, ED, MM);
    edge_exp_kernel<<<blocks((long long)Etot * 2, BT), BT, 0, stream>>>(
        ei, E, Etot, ES, ED, MM, SS, P);
    edge_norm_kernel<<<blocks((long long)Etot * 2, BT), BT, 0, stream>>>(
        ei, E, Etot, SS, P);
    edge_aggr_kernel<<<(unsigned)Etot, 256, 0, stream>>>(ei, E, P, HB, XB);

    bias_relu_kernel<<<blocks((long long)NN * HC, BT), BT, 0, stream>>>(
        XB, bc, NN * HC);

    xin = XB; F = HC; Kpad = KP2;
  }

  // global mean pool + MLP head
  fill_f32<<<blocks(GG * HC, BT), BT, 0, stream>>>(POOL, 0.f, GG * HC);
  fill_f32<<<blocks(GG, BT), BT, 0, stream>>>(CNT, 0.f, GG);
  pool_kernel<<<blocks((long long)NN * HC, BT), BT, 0, stream>>>(XB, batch, POOL, CNT);

  const float* lw1 = (const float*)d_in[23]; const float* lb1 = (const float*)d_in[24];
  const float* lw2 = (const float*)d_in[25]; const float* lb2 = (const float*)d_in[26];
  const float* lw3 = (const float*)d_in[27]; const float* lb3 = (const float*)d_in[28];

  mlp_kernel<<<GG, 128, 0, stream>>>(POOL, lw1, lb1, M1, 200, 100, 1, CNT);
  mlp_kernel<<<GG, 128, 0, stream>>>(M1,   lw2, lb2, M2, 100, 100, 1, nullptr);
  mlp_kernel<<<GG, 128, 0, stream>>>(M2,   lw3, lb3, (float*)d_out, 100, 29, 0, nullptr);
}
